// SparseAttention_69020124446895
// MI455X (gfx1250) — matrix operand order repaired and checked by reading, not compiled, not run on hardware
//
#include <hip/hip_runtime.h>
#include <hip/hip_bf16.h>

// ---------------------------------------------------------------------------
// SparseAttention on MI455X (gfx1250): bf16 WMMA everywhere.
//   B=2, S=4096, D=512, H=8, HD=64
//   mask = (causal && same 128-block) || (causal && (k%128==0 || k%128>=96))
// ---------------------------------------------------------------------------

typedef __attribute__((ext_vector_type(16))) __bf16 v16bf;
typedef __attribute__((ext_vector_type(8)))  __bf16 v8bf;
typedef __attribute__((ext_vector_type(8)))  float  v8f;

union FragBF { v16bf v; v8bf h[2]; __bf16 e[16]; };

__device__ __forceinline__ v8f vzero8() {
    v8f z = {0.f,0.f,0.f,0.f,0.f,0.f,0.f,0.f};
    return z;
}
__device__ __forceinline__ void wait_ds() {
    asm volatile("s_wait_dscnt 0x0" ::: "memory");
}
__device__ __forceinline__ int imin(int a, int b) { return a < b ? a : b; }

#define WMMA_BF16(A, B, C) \
    __builtin_amdgcn_wmma_f32_16x16x32_bf16(false, (A), false, (B), (short)0, (C), false, false)

// A-operand (16x32 bf16, M=row per lane): frag[e]=K koff+e (contiguous), koff=0/16 per half-wave
// B-operand (32x16 bf16, N=col per lane): frag[e]=K koff+e (contiguous), koff=0/16 per half-wave

static constexpr int S_LEN = 4096;
static constexpr int DMODEL = 512;
static constexpr int HEADS = 8;
static constexpr int HD = 64;
static constexpr int BATCH = 2;

// ---------------------------------------------------------------------------
// Kernel 1: 512x512 transpose, f32 -> bf16  (dst[n][k] = src[k][n])
// ---------------------------------------------------------------------------
__global__ __launch_bounds__(256) void k_transpose512(const float* __restrict__ src,
                                                      __bf16* __restrict__ dst) {
    int idx = blockIdx.x * 256 + threadIdx.x;      // 512*512 elements
    int n = idx >> 9, k = idx & 511;
    dst[idx] = (__bf16)src[k * 512 + n];
}

// load 16x32 A-fragment half-rows from an f32 row, converting to bf16
__device__ __forceinline__ v16bf load_a_f32(const float* __restrict__ rowp, int base) {
    FragBF f;
#pragma unroll
    for (int i = 0; i < 8; ++i) f.e[i]     = (__bf16)rowp[base + i];
#pragma unroll
    for (int i = 0; i < 8; ++i) f.e[8 + i] = (__bf16)rowp[base + 8 + i];
    return f.v;
}

// ---------------------------------------------------------------------------
// Kernel 2: fused QKV projection.  M=B*S=8192, N=512, K=512.
//   z=0 -> Q [B,H,S,HD] bf16, scaled by 1/8
//   z=1 -> K [B,H,S,HD] bf16
//   z=2 -> V transposed [B,H,HD,S] bf16
// Block = 8 waves, each wave = 16 rows x 64 cols.
// ---------------------------------------------------------------------------
__global__ __launch_bounds__(256) void k_qkv_gemm(const float* __restrict__ X,
                                                  const __bf16* __restrict__ WtQ,
                                                  const __bf16* __restrict__ WtK,
                                                  const __bf16* __restrict__ WtV,
                                                  const float* __restrict__ bq,
                                                  const float* __restrict__ bk,
                                                  const float* __restrict__ bv,
                                                  __bf16* __restrict__ Qo,
                                                  __bf16* __restrict__ Ko,
                                                  __bf16* __restrict__ Vt) {
    const int z = blockIdx.z;
    const __bf16* Wt  = (z == 0) ? WtQ : (z == 1) ? WtK : WtV;
    const float* bias = (z == 0) ? bq  : (z == 1) ? bk  : bv;

    const int w    = threadIdx.x >> 5;
    const int lane = threadIdx.x & 31;
    const int m0   = blockIdx.x * 128 + w * 16;
    const int n0   = blockIdx.y * 64;
    const int r    = lane & 15;
    const int b8   = (lane < 16) ? 0 : 8;
    const int koff = (lane < 16) ? 0 : 16;

    const float* arow = X + (size_t)(m0 + r) * DMODEL;

    v8f acc[4];
#pragma unroll
    for (int t = 0; t < 4; ++t) acc[t] = vzero8();

    for (int kc = 0; kc < DMODEL; kc += 32) {
        v16bf a = load_a_f32(arow, kc + koff);
#pragma unroll
        for (int t = 0; t < 4; ++t) {
            const __bf16* wrow = Wt + (size_t)(n0 + t * 16 + r) * DMODEL;
            v16bf bb = *(const v16bf*)(wrow + kc + koff);
            acc[t] = WMMA_BF16(a, bb, acc[t]);
        }
    }

    const float scale = (z == 0) ? 0.125f : 1.0f;  // 1/sqrt(64) folded into Q
#pragma unroll
    for (int t = 0; t < 4; ++t) {
        int col = n0 + t * 16 + r;
        float bb = bias[col];
        int h = col >> 6, hd = col & 63;
#pragma unroll
        for (int vr = 0; vr < 8; ++vr) {
            int m = m0 + vr + b8;
            int b = m >> 12, s = m & 4095;
            float val = (acc[t][vr] + bb) * scale;
            if (z == 0)
                Qo[(((size_t)b * HEADS + h) * S_LEN + s) * HD + hd] = (__bf16)val;
            else if (z == 1)
                Ko[(((size_t)b * HEADS + h) * S_LEN + s) * HD + hd] = (__bf16)val;
            else
                Vt[(((size_t)b * HEADS + h) * HD + hd) * S_LEN + s] = (__bf16)val;
        }
    }
}

// ---------------------------------------------------------------------------
// Kernel 3: sparse flash attention.
// grid = (S/128, H, B); block = 8 waves; wave w handles 16 queries.
// Key tiles of 32:
//   kind 0: diagonal block tiles (causal mask inside)
//   kind 1: earlier-block cols 96..127 (always valid)
//   kind 2: stride-128 "col 0" columns, gathered 32 at a time
// ---------------------------------------------------------------------------
__global__ __launch_bounds__(256) void k_attn(const __bf16* __restrict__ Q,
                                              const __bf16* __restrict__ K,
                                              const __bf16* __restrict__ Vt,
                                              __bf16* __restrict__ Xo) {
    __shared__ __align__(16) float  sS[8][16][32];
    __shared__ __align__(16) __bf16 sP[8][16][32];
    __shared__ __align__(16) float  sFac[8][16];

    const int qb = blockIdx.x, h = blockIdx.y, b = blockIdx.z;
    const int w    = threadIdx.x >> 5;
    const int lane = threadIdx.x & 31;
    const int r    = lane & 15;
    const int b8   = (lane < 16) ? 0 : 8;
    const int koff = (lane < 16) ? 0 : 16;
    const int qbase = qb * 128 + w * 16;
    const int q     = qbase + r;

    const size_t headoff = ((size_t)b * HEADS + h) * S_LEN * HD;
    const __bf16* Qh = Q  + headoff;   // [S][64]
    const __bf16* Kh = K  + headoff;   // [S][64]
    const __bf16* Vh = Vt + headoff;   // [64][S]

    float (*Sw)[32]  = sS[w];
    __bf16 (*Pw)[32] = sP[w];
    float* facw      = sFac[w];

    // Q fragments (reused across all key tiles): hd 0..31 and 32..63
    FragBF aq0, aq1;
    {
        const __bf16* qrow = Qh + (size_t)q * HD;
        aq0.h[0] = *(const v8bf*)(qrow + koff);
        aq0.h[1] = *(const v8bf*)(qrow + koff + 8);
        aq1.h[0] = *(const v8bf*)(qrow + 32 + koff);
        aq1.h[1] = *(const v8bf*)(qrow + 32 + koff + 8);
    }

    v8f accO[4];
#pragma unroll
    for (int t = 0; t < 4; ++t) accO[t] = vzero8();
    float run_max = -1e30f, run_sum = 0.0f;

    auto process = [&](int kind, int kb, int g0) {
        // ---- logits: two 16-key sub-tiles, each 2 WMMAs over hd=64 ----
        v8f sAcc[2];
#pragma unroll
        for (int sub = 0; sub < 2; ++sub) {
            int key;
            if (kind == 2) key = imin(g0 + sub * 16 + r, qb - 1) * 128;
            else           key = kb + sub * 16 + r;
            const __bf16* krow = Kh + (size_t)key * HD;
            v16bf bk0 = *(const v16bf*)(krow + koff);
            v16bf bk1 = *(const v16bf*)(krow + 32 + koff);
            v8f sc = vzero8();
            sc = WMMA_BF16(aq0.v, bk0, sc);
            sc = WMMA_BF16(aq1.v, bk1, sc);
            sAcc[sub] = sc;
        }
        // ---- C-layout -> LDS (row-major 16x32 f32) ----
#pragma unroll
        for (int vr = 0; vr < 8; ++vr) {
            int row = vr + b8;
            Sw[row][r]      = sAcc[0][vr];
            Sw[row][r + 16] = sAcc[1][vr];
        }
        wait_ds();
        // ---- masked online softmax (lanes 0..15 own one query row each) ----
        if (lane < 16) {
            float sv[32];
#pragma unroll
            for (int j = 0; j < 32; ++j) {
                bool valid;
                if (kind == 0)      valid = (kb + j) <= q;
                else if (kind == 1) valid = true;
                else                valid = (g0 + j) < qb;
                sv[j] = valid ? Sw[r][j] : -1e10f;
            }
            float tm = sv[0];
#pragma unroll
            for (int j = 1; j < 32; ++j) tm = fmaxf(tm, sv[j]);
            float nm   = fmaxf(run_max, tm);
            float corr = __expf(run_max - nm);
            float sum  = 0.f;
#pragma unroll
            for (int j = 0; j < 32; ++j) {
                float p = __expf(sv[j] - nm);
                sum += p;
                Pw[r][j] = (__bf16)p;
            }
            run_sum = run_sum * corr + sum;
            run_max = nm;
            facw[r] = corr;
        }
        wait_ds();
        // ---- rescale running O accumulators ----
#pragma unroll
        for (int vr = 0; vr < 8; ++vr) {
            float cf = facw[vr + b8];
#pragma unroll
            for (int t = 0; t < 4; ++t) accO[t][vr] *= cf;
        }
        // ---- P (A-operand) from LDS ----
        FragBF ap;
        ap.h[0] = *(const v8bf*)&Pw[r][koff];
        ap.h[1] = *(const v8bf*)&Pw[r][koff + 8];
        // ---- O += P * V  (V^T rows are contiguous in keys) ----
#pragma unroll
        for (int t = 0; t < 4; ++t) {
            const __bf16* vrow = Vh + (size_t)(t * 16 + r) * S_LEN;
            v16bf bv;
            if (kind != 2) {
                bv = *(const v16bf*)(vrow + kb + koff);
            } else {
                FragBF fv;
#pragma unroll
                for (int e = 0; e < 16; ++e) {
                    int idx = imin(g0 + koff + e, qb - 1);
                    fv.e[e] = vrow[idx * 128];
                }
                bv = fv.v;
            }
            accO[t] = WMMA_BF16(ap.v, bv, accO[t]);
        }
    };

    // diagonal block (per-wave pruned trip count; waves are independent)
    const int tmaxd = (w * 16 + 15) >> 5;
    for (int t = 0; t <= tmaxd; ++t) process(0, qb * 128 + t * 32, 0);
    // earlier blocks: cols 96..127
    for (int j = 0; j < qb; ++j) process(1, j * 128 + 96, 0);
    // earlier blocks: col 0 (stride 128), gathered 32 at a time
    for (int g0 = 0; g0 < qb; g0 += 32) process(2, 0, g0);

    // ---- normalize and store x -> [B, S, H*HD] bf16 ----
    if (lane < 16) facw[r] = 1.0f / run_sum;
    wait_ds();
#pragma unroll
    for (int vr = 0; vr < 8; ++vr) {
        int m = vr + b8;
        float inv = facw[m];
        int s = qbase + m;
#pragma unroll
        for (int t = 0; t < 4; ++t) {
            Xo[((size_t)b * S_LEN + s) * DMODEL + h * HD + t * 16 + r] =
                (__bf16)(accO[t][vr] * inv);
        }
    }
}

// ---------------------------------------------------------------------------
// Kernel 4: output projection. out = X[8192,512] * Wot^T + bo, f32 result.
// ---------------------------------------------------------------------------
__global__ __launch_bounds__(256) void k_out_gemm(const __bf16* __restrict__ X,
                                                  const __bf16* __restrict__ Wot,
                                                  const float* __restrict__ bo,
                                                  float* __restrict__ out) {
    const int w    = threadIdx.x >> 5;
    const int lane = threadIdx.x & 31;
    const int m0   = blockIdx.x * 128 + w * 16;
    const int n0   = blockIdx.y * 64;
    const int r    = lane & 15;
    const int b8   = (lane < 16) ? 0 : 8;
    const int koff = (lane < 16) ? 0 : 16;

    const __bf16* arow = X + (size_t)(m0 + r) * DMODEL;

    v8f acc[4];
#pragma unroll
    for (int t = 0; t < 4; ++t) acc[t] = vzero8();

    for (int kc = 0; kc < DMODEL; kc += 32) {
        FragBF a;
        a.h[0] = *(const v8bf*)(arow + kc + koff);
        a.h[1] = *(const v8bf*)(arow + kc + koff + 8);
#pragma unroll
        for (int t = 0; t < 4; ++t) {
            const __bf16* wrow = Wot + (size_t)(n0 + t * 16 + r) * DMODEL;
            v16bf bb = *(const v16bf*)(wrow + kc + koff);
            acc[t] = WMMA_BF16(a.v, bb, acc[t]);
        }
    }

#pragma unroll
    for (int t = 0; t < 4; ++t) {
        int col = n0 + t * 16 + r;
        float bias = bo[col];
#pragma unroll
        for (int vr = 0; vr < 8; ++vr) {
            int m = m0 + vr + b8;
            out[(size_t)m * DMODEL + col] = acc[t][vr] + bias;
        }
    }
}

// ---------------------------------------------------------------------------
// Host-side launch
// ---------------------------------------------------------------------------
extern "C" void kernel_launch(void* const* d_in, const int* in_sizes, int n_in,
                              void* d_out, int out_size, void* d_ws, size_t ws_size,
                              hipStream_t stream) {
    const float* x  = (const float*)d_in[0];
    const float* wq = (const float*)d_in[1];
    const float* bq = (const float*)d_in[2];
    const float* wk = (const float*)d_in[3];
    const float* bk = (const float*)d_in[4];
    const float* wv = (const float*)d_in[5];
    const float* bv = (const float*)d_in[6];
    const float* wo = (const float*)d_in[7];
    const float* bo = (const float*)d_in[8];

    __bf16* wsb = (__bf16*)d_ws;
    const size_t WSZ  = 512 * 512;                       // one transposed weight
    const size_t QKSZ = (size_t)BATCH * HEADS * S_LEN * HD;
    __bf16* WtQ = wsb;
    __bf16* WtK = WtQ + WSZ;
    __bf16* WtV = WtK + WSZ;
    __bf16* Wot = WtV + WSZ;
    __bf16* Qb  = Wot + WSZ;
    __bf16* Kb  = Qb + QKSZ;
    __bf16* Vtb = Kb + QKSZ;
    __bf16* Xb  = Vtb + QKSZ;                            // [B,S,512]

    // 1) transpose + downconvert weights (B-operand rows get contiguous K)
    k_transpose512<<<1024, 256, 0, stream>>>(wq, WtQ);
    k_transpose512<<<1024, 256, 0, stream>>>(wk, WtK);
    k_transpose512<<<1024, 256, 0, stream>>>(wv, WtV);
    k_transpose512<<<1024, 256, 0, stream>>>(wo, Wot);

    // 2) fused QKV projection (z selects q/k/v)
    k_qkv_gemm<<<dim3(64, 8, 3), 256, 0, stream>>>(x, WtQ, WtK, WtV, bq, bk, bv,
                                                   Qb, Kb, Vtb);

    // 3) sparse flash attention
    k_attn<<<dim3(S_LEN / 128, HEADS, BATCH), 256, 0, stream>>>(Qb, Kb, Vtb, Xb);

    // 4) output projection
    k_out_gemm<<<dim3(64, 8), 256, 0, stream>>>(Xb, Wot, bo, (float*)d_out);
}
